// CConv_39359080300789
// MI455X (gfx1250) — matrix operand order repaired and checked by reading, not compiled, not run on hardware
//
#include <hip/hip_runtime.h>
#include <hip/hip_bf16.h>

typedef __attribute__((ext_vector_type(16))) _Float16 v16h;
typedef __attribute__((ext_vector_type(8)))  _Float16 v8h;
typedef __attribute__((ext_vector_type(8)))  float    v8f;

#define OC 128
#define IC 128
#define H  112
#define W  112
#define KTAPS 9

// LDS x-tile layout: [ih(4)][col(114)][ic(128)] f16, col stride padded to 136 f16
#define LDS_C_STRIDE 136
#define LDS_IH_STRIDE (114 * LDS_C_STRIDE)

// ---------------------------------------------------------------------------
// Kernel 1: dequantize codebook weights into f16, layout [r][oc][ic]
// weight = centroids[idx[oc*IC+ic]][r] * fp16_roundtrip(scales) * cut
// ---------------------------------------------------------------------------
__global__ void cconv_dequant(const float* __restrict__ centroids,
                              const int*   __restrict__ idx,
                              const float* __restrict__ scales,
                              const float* __restrict__ cut,
                              _Float16*    __restrict__ wq)
{
    int t = blockIdx.x * 256 + threadIdx.x;       // 0 .. OC*IC-1
    if (t >= OC * IC) return;
    int oc = t / IC;
    int ic = t - oc * IC;
    float s  = (float)(_Float16)scales[t];        // fp16 round-trip (RTNE)
    float cu = cut[t];
    const float* c9 = centroids + (size_t)idx[t] * KTAPS;
#pragma unroll
    for (int r = 0; r < KTAPS; r++) {
        wq[((size_t)r * OC + oc) * IC + ic] = (_Float16)(c9[r] * s * cu);
    }
}

// ---------------------------------------------------------------------------
// Kernel 2: implicit-GEMM conv. Block = 512 thr = 16 waves.
// wave = (mtile 0..7 [16 oc each], row 0..1). Block covers (n, oh0..oh0+1),
// full 112-wide output row per wave as 7 x 16x16 WMMA tiles, K = IC*9 = 1152.
// ---------------------------------------------------------------------------
__global__ __launch_bounds__(512) void cconv_wmma(
    const float*    __restrict__ x,
    const _Float16* __restrict__ wq,
    const float*    __restrict__ bias,
    float*          __restrict__ out)
{
    __shared__ __align__(16) _Float16 xs[4 * LDS_IH_STRIDE];

    const int n   = blockIdx.y;      // batch 0..15
    const int oh0 = blockIdx.x * 2;  // output row pair 0..110
    const int tid = threadIdx.x;

    // ---- stage x halo tile (ih = oh0-1 .. oh0+2, f32 -> f16) into LDS ----
    const int ROW_ELEMS = IC * W;    // 14336 per ih row
    for (int e = tid; e < 4 * ROW_ELEMS; e += 512) {
        int ihl = e / ROW_ELEMS;
        int rem = e - ihl * ROW_ELEMS;
        int ic  = rem / W;
        int iw  = rem - ic * W;
        int ihg = oh0 - 1 + ihl;
        float v = 0.0f;
        if (ihg >= 0 && ihg < H)
            v = x[(((size_t)n * IC + ic) * H + ihg) * W + iw];
        xs[ihl * LDS_IH_STRIDE + (iw + 1) * LDS_C_STRIDE + ic] = (_Float16)v;
    }
    // zero-pad border columns (col 0 and col 113)
    for (int e = tid; e < 4 * IC * 2; e += 512) {
        int ihl = e >> 8;            // / 256
        int rem = e & 255;
        int ic  = rem >> 1;
        int c   = (rem & 1) ? 113 : 0;
        xs[ihl * LDS_IH_STRIDE + c * LDS_C_STRIDE + ic] = (_Float16)0.0f;
    }
    __syncthreads();

    const int wave = tid >> 5;       // wave32
    const int lane = tid & 31;
    const int mt   = wave & 7;       // oc tile
    const int row  = wave >> 3;      // which output row of the pair
    const int oh   = oh0 + row;
    const int lo   = lane & 15;
    const int hi   = lane >> 4;

    // A-matrix row this lane supplies (lanes 0-15 and 16-31 both cover M=0..15)
    const int ocA = mt * 16 + lo;

    // ---- accumulators, init with bias (C layout: VGPR v -> M = v + 8*hi) ----
    v8f acc[7];
#pragma unroll
    for (int j = 0; j < 7; j++)
#pragma unroll
        for (int v = 0; v < 8; v++)
            acc[j][v] = bias[mt * 16 + hi * 8 + v];

    // ---- K loop: 9 taps x 4 ic-blocks of 32 ----
    for (int r = 0; r < KTAPS; r++) {
        const int kh = r / 3;
        const int kw = r - kh * 3;
        const _Float16* wrow = wq + ((size_t)r * OC + ocA) * IC;
        const _Float16* xrow = xs + (row + kh) * LDS_IH_STRIDE + kw * LDS_C_STRIDE;
#pragma unroll
        for (int icb = 0; icb < 4; icb++) {
            // A frag: lanes 0-15: v0-3 = K0..7, v4-7 = K16..23 (hi lanes +8)
            const int kA = icb * 32 + hi * 8;
            v8h a0 = *(const v8h*)(wrow + kA);
            v8h a1 = *(const v8h*)(wrow + kA + 16);
            v16h A;
#pragma unroll
            for (int i = 0; i < 8; i++) { A[i] = a0[i]; A[8 + i] = a1[i]; }

            const int kB = icb * 32 + hi * 8;    // B: K0..7 then K16..23 (hi lanes +8), same as A
#pragma unroll
            for (int j = 0; j < 7; j++) {
                const _Float16* bp = xrow + (j * 16 + lo) * LDS_C_STRIDE + kB;
                v8h b0 = *(const v8h*)(bp);
                v8h b1 = *(const v8h*)(bp + 16);
                v16h B;
#pragma unroll
                for (int i = 0; i < 8; i++) { B[i] = b0[i]; B[8 + i] = b1[i]; }

                acc[j] = __builtin_amdgcn_wmma_f32_16x16x32_f16(
                    /*neg_a=*/false, A, /*neg_b=*/false, B,
                    /*c_mod=*/(short)0, acc[j],
                    /*reuse_a=*/false, /*reuse_b=*/false);
            }
        }
    }

    // ---- store: D layout VGPR v -> M = v + 8*hi, N = lo ----
#pragma unroll
    for (int j = 0; j < 7; j++) {
        const int ow = j * 16 + lo;
#pragma unroll
        for (int v = 0; v < 8; v++) {
            const int oc = mt * 16 + hi * 8 + v;
            out[(((size_t)n * OC + oc) * H + oh) * W + ow] = acc[j][v];
        }
    }
}

extern "C" void kernel_launch(void* const* d_in, const int* in_sizes, int n_in,
                              void* d_out, int out_size, void* d_ws, size_t ws_size,
                              hipStream_t stream) {
    const float* x         = (const float*)d_in[0];
    const float* centroids = (const float*)d_in[1];
    const int*   idx       = (const int*)  d_in[2];
    const float* scales    = (const float*)d_in[3];
    const float* cut       = (const float*)d_in[4];
    const float* bias      = (const float*)d_in[5];
    float*       out       = (float*)d_out;
    _Float16*    wq        = (_Float16*)d_ws;   // 9*128*128 f16 = 294,912 B

    // 1) dequantize weights into workspace
    cconv_dequant<<<(OC * IC + 255) / 256, 256, 0, stream>>>(
        centroids, idx, scales, cut, wq);

    // 2) implicit-GEMM conv: grid = (56 oh-pairs, 16 batches)
    dim3 grid(H / 2, 16);
    cconv_wmma<<<grid, 512, 0, stream>>>(x, wq, bias, out);
}
